// ClusterNetHetero_35356170780710
// MI455X (gfx1250) — compile-verified, matrix operand order checked by reading, not run on hardware
//
#include <hip/hip_runtime.h>
#include <math.h>

// ---------------- problem constants ----------------
#define N_NODES 65536
#define N_EDGES 1048576
#define NB      32
#define D       64
#define NL      4
#define NC      2

typedef __attribute__((ext_vector_type(16))) _Float16 v16h;
typedef __attribute__((ext_vector_type(8)))  float    v8f;

// 16-bit A/B fragment K index for half-element h (0..15) and lane-half hi (0/1)
// per CDNA5 ISA 7.12.2: lanes 0-15 hold K {0..7,16..23}, lanes 16-31 K {8..15,24..31}
__device__ __forceinline__ int kidx16(int h, int hi) {
    return (h < 8 ? h : h + 8) + 8 * hi;
}

__device__ __forceinline__ v8f wmma_f16(v16h a, v16h b, v8f c) {
    return __builtin_amdgcn_wmma_f32_16x16x32_f16(
        /*neg_a=*/false, a, /*neg_b=*/false, b,
        /*c_mod=*/(short)0, c, /*reuse_a=*/false, /*reuse_b=*/false);
}

// B fragment from pre-swizzled weights: one contiguous 32B per lane
// layout: [((ch*4 + nt)*32 + lane)*16 + h]
__device__ __forceinline__ v16h load_bfrag(const _Float16* __restrict__ Wsw,
                                           int ch, int nt, int lane) {
    return *(const v16h*)(Wsw + ((((ch * 4 + nt) * 32) + lane) << 4));
}

// A fragment from an f32 row (global or LDS): two contiguous 8-float runs per chunk
__device__ __forceinline__ v16h pack_afrag(const float* __restrict__ row, int base, int hi) {
    const float4* r4 = (const float4*)row;
    int i0 = (base + 8 * hi) >> 2;
    int i1 = (base + 16 + 8 * hi) >> 2;
    float4 q0 = r4[i0], q1 = r4[i0 + 1], q2 = r4[i1], q3 = r4[i1 + 1];
    v16h a;
    a[0]  = (_Float16)q0.x; a[1]  = (_Float16)q0.y; a[2]  = (_Float16)q0.z; a[3]  = (_Float16)q0.w;
    a[4]  = (_Float16)q1.x; a[5]  = (_Float16)q1.y; a[6]  = (_Float16)q1.z; a[7]  = (_Float16)q1.w;
    a[8]  = (_Float16)q2.x; a[9]  = (_Float16)q2.y; a[10] = (_Float16)q2.z; a[11] = (_Float16)q2.w;
    a[12] = (_Float16)q3.x; a[13] = (_Float16)q3.y; a[14] = (_Float16)q3.z; a[15] = (_Float16)q3.w;
    return a;
}

// ordered-int trick float atomic max (buffer initialized to -inf)
__device__ __forceinline__ void atomicMaxFloat(float* addr, float v) {
    if (v >= 0.0f) atomicMax((int*)addr, __float_as_int(v));
    else           atomicMin((unsigned int*)addr, __float_as_uint(v));
}

// ---------------- weight f32 -> f16 with B-fragment swizzle ----------------
// dst[((L*8 + ch*4 + nt)*32 + lane)*16 + h] = src[L*4096 + (ch*32 + kidx16(h,hi))*64 + nt*16 + p]
__global__ void swizzle_weights(const float* __restrict__ src, _Float16* __restrict__ dst) {
    int i = blockIdx.x * blockDim.x + threadIdx.x;   // over NL*4096
    int h    = i & 15;
    int lane = (i >> 4) & 31;
    int nt   = (i >> 9) & 3;
    int ch   = (i >> 11) & 1;
    int L    = i >> 12;
    int p = lane & 15, hi = lane >> 4;
    int k = ch * 32 + kidx16(h, hi);
    dst[i] = (_Float16)src[(size_t)L * 4096 + (size_t)k * D + nt * 16 + p];
}

// ---------------- node GEMMs: v / a_src / a_dst ----------------
__device__ __forceinline__ void gemm_one(const _Float16* __restrict__ Wsw,
                                         const v16h& a0, const v16h& a1,
                                         float* __restrict__ out,
                                         int row0, int p, int hi, int lane) {
#pragma unroll
    for (int nt = 0; nt < 4; ++nt) {
        v8f c = {};
        c = wmma_f16(a0, load_bfrag(Wsw, 0, nt, lane), c);
        c = wmma_f16(a1, load_bfrag(Wsw, 1, nt, lane), c);
#pragma unroll
        for (int r = 0; r < 8; ++r) {
            int m = r + 8 * hi;
            out[(size_t)(row0 + m) * D + nt * 16 + p] = c[r];
        }
    }
}

__global__ void node_gemm(const float* __restrict__ h,
                          const _Float16* __restrict__ Wl,
                          const _Float16* __restrict__ Ws,
                          const _Float16* __restrict__ Wd,
                          float* __restrict__ vout,
                          float* __restrict__ asout,
                          float* __restrict__ adout) {
    int lane = threadIdx.x & 31;
    int wave = threadIdx.x >> 5;
    int tile = blockIdx.x * 8 + wave;     // 4096 tiles of 16 rows
    int row0 = tile * 16;
    int p = lane & 15, hi = lane >> 4;

    const float* row = h + (size_t)(row0 + p) * D;
    v16h a0 = pack_afrag(row, 0, hi);
    v16h a1 = pack_afrag(row, 32, hi);

    gemm_one(Wl, a0, a1, vout,  row0, p, hi, lane);
    gemm_one(Ws, a0, a1, asout, row0, p, hi, lane);
    gemm_one(Wd, a0, a1, adout, row0, p, hi, lane);
}

// ---------------- edge pass B: delta + alpha_pre + segment max ----------------
__global__ void edge_pass_b(const int* __restrict__ srcIdx, const int* __restrict__ dstIdx,
                            const float* __restrict__ pos,
                            const float* __restrict__ asbuf, const float* __restrict__ adbuf,
                            const float* __restrict__ pW1, const float* __restrict__ pb1,
                            const _Float16* __restrict__ pW2, const float* __restrict__ pb2,
                            const _Float16* __restrict__ aW1, const float* __restrict__ ab1,
                            const _Float16* __restrict__ aW2, const float* __restrict__ ab2,
                            float* __restrict__ deltaWs, float* __restrict__ alphaWs,
                            float* __restrict__ mbuf) {
    __shared__ float lds[8][16][68];   // per-wave 16x64 transpose staging (16B-aligned rows)
    int lane = threadIdx.x & 31;
    int wave = threadIdx.x >> 5;
    int tile = blockIdx.x * 8 + wave;  // 65536 tiles of 16 edges
    int e0 = tile * 16;
    int p = lane & 15, hi = lane >> 4;

    int s = srcIdx[e0 + p];
    int d = dstIdx[e0 + p];
    float px = pos[(size_t)d * 2]     - pos[(size_t)s * 2];
    float py = pos[(size_t)d * 2 + 1] - pos[(size_t)s * 2 + 1];

    // hidden1 = relu(pdiff @ pW1 + pb1), produced directly in A-fragment layout
    v16h a0, a1;
#pragma unroll
    for (int hh = 0; hh < 16; ++hh) {
        int k = kidx16(hh, hi);
        float c0 = fmaxf(0.0f, fmaf(px, pW1[k],      fmaf(py, pW1[D + k],      pb1[k])));
        float c1 = fmaxf(0.0f, fmaf(px, pW1[32 + k], fmaf(py, pW1[D + 32 + k], pb1[32 + k])));
        a0[hh] = (_Float16)c0;
        a1[hh] = (_Float16)c1;
    }

    // delta = relu(hidden1 @ pW2 + pb2)   (C layout: row m=r+8*hi, col nt*16+p)
    v8f deltaC[4];
#pragma unroll
    for (int nt = 0; nt < 4; ++nt) {
        v8f c;
        float bias = pb2[nt * 16 + p];
#pragma unroll
        for (int r = 0; r < 8; ++r) c[r] = bias;
        c = wmma_f16(a0, load_bfrag(pW2, 0, nt, lane), c);
        c = wmma_f16(a1, load_bfrag(pW2, 1, nt, lane), c);
#pragma unroll
        for (int r = 0; r < 8; ++r) c[r] = fmaxf(c[r], 0.0f);
        deltaC[nt] = c;
        float* dptr = deltaWs + (size_t)tile * 1024 + nt * 256 + lane * 8;
        *(float4*)(dptr)     = make_float4(c[0], c[1], c[2], c[3]);
        *(float4*)(dptr + 4) = make_float4(c[4], c[5], c[6], c[7]);
    }

    // per-C-row edge ids via wave shuffle (lane m holds src/dst of edge e0+m)
    int sm[8], dm[8];
#pragma unroll
    for (int r = 0; r < 8; ++r) {
        int m = r + 8 * hi;
        sm[r] = __shfl(s, m, 32);
        dm[r] = __shfl(d, m, 32);
    }

    // t = a_dst[dst] - a_src[src] + delta  -> stage into LDS for transpose
#pragma unroll
    for (int nt = 0; nt < 4; ++nt) {
        int cch = nt * 16 + p;
#pragma unroll
        for (int r = 0; r < 8; ++r) {
            int m = r + 8 * hi;
            float t = adbuf[(size_t)dm[r] * D + cch] - asbuf[(size_t)sm[r] * D + cch] + deltaC[nt][r];
            lds[wave][m][cch] = t;
        }
    }
    asm volatile("s_wait_dscnt 0x0" ::: "memory");

    v16h ta0 = pack_afrag(&lds[wave][p][0], 0, hi);
    v16h ta1 = pack_afrag(&lds[wave][p][0], 32, hi);

    // hidden2 = relu(t @ aW1 + ab1)
    v8f h2[4];
#pragma unroll
    for (int nt = 0; nt < 4; ++nt) {
        v8f c;
        float bias = ab1[nt * 16 + p];
#pragma unroll
        for (int r = 0; r < 8; ++r) c[r] = bias;
        c = wmma_f16(ta0, load_bfrag(aW1, 0, nt, lane), c);
        c = wmma_f16(ta1, load_bfrag(aW1, 1, nt, lane), c);
#pragma unroll
        for (int r = 0; r < 8; ++r) c[r] = fmaxf(c[r], 0.0f);
        h2[nt] = c;
    }
    asm volatile("s_wait_dscnt 0x0" ::: "memory");
#pragma unroll
    for (int nt = 0; nt < 4; ++nt)
#pragma unroll
        for (int r = 0; r < 8; ++r)
            lds[wave][r + 8 * hi][nt * 16 + p] = h2[nt][r];
    asm volatile("s_wait_dscnt 0x0" ::: "memory");

    v16h ha0 = pack_afrag(&lds[wave][p][0], 0, hi);
    v16h ha1 = pack_afrag(&lds[wave][p][0], 32, hi);

    // alpha_pre = relu(hidden2 @ aW2 + ab2); store + segment atomic max
#pragma unroll
    for (int nt = 0; nt < 4; ++nt) {
        v8f c;
        float bias = ab2[nt * 16 + p];
#pragma unroll
        for (int r = 0; r < 8; ++r) c[r] = bias;
        c = wmma_f16(ha0, load_bfrag(aW2, 0, nt, lane), c);
        c = wmma_f16(ha1, load_bfrag(aW2, 1, nt, lane), c);
        int cch = nt * 16 + p;
#pragma unroll
        for (int r = 0; r < 8; ++r) c[r] = fmaxf(c[r], 0.0f);
        float* aptr = alphaWs + (size_t)tile * 1024 + nt * 256 + lane * 8;
        *(float4*)(aptr)     = make_float4(c[0], c[1], c[2], c[3]);
        *(float4*)(aptr + 4) = make_float4(c[4], c[5], c[6], c[7]);
#pragma unroll
        for (int r = 0; r < 8; ++r)
            atomicMaxFloat(&mbuf[(size_t)dm[r] * D + cch], c[r]);
    }
}

// ---------------- edge pass C: exp + num/den accumulation ----------------
__global__ void edge_pass_c(const int* __restrict__ srcIdx, const int* __restrict__ dstIdx,
                            const float* __restrict__ vbuf, const float* __restrict__ mbuf,
                            const float* __restrict__ deltaWs, const float* __restrict__ alphaWs,
                            float* __restrict__ numb, float* __restrict__ denb) {
    int lane = threadIdx.x & 31;
    int wave = threadIdx.x >> 5;
    int tile = blockIdx.x * 8 + wave;
    int e0 = tile * 16;
    int p = lane & 15, hi = lane >> 4;

    int s = srcIdx[e0 + p];
    int d = dstIdx[e0 + p];
    int sm[8], dm[8];
#pragma unroll
    for (int r = 0; r < 8; ++r) {
        int m = r + 8 * hi;
        sm[r] = __shfl(s, m, 32);
        dm[r] = __shfl(d, m, 32);
    }
#pragma unroll
    for (int nt = 0; nt < 4; ++nt) {
        int cch = nt * 16 + p;
        const float4* ap = (const float4*)(alphaWs + (size_t)tile * 1024 + nt * 256 + lane * 8);
        const float4* dp = (const float4*)(deltaWs + (size_t)tile * 1024 + nt * 256 + lane * 8);
        float4 a0 = ap[0], a1 = ap[1];
        float4 d0 = dp[0], d1 = dp[1];
        float al[8] = { a0.x, a0.y, a0.z, a0.w, a1.x, a1.y, a1.z, a1.w };
        float de[8] = { d0.x, d0.y, d0.z, d0.w, d1.x, d1.y, d1.z, d1.w };
#pragma unroll
        for (int r = 0; r < 8; ++r) {
            float mg = mbuf[(size_t)dm[r] * D + cch];
            float ex = __expf(al[r] - mg);
            float vg = vbuf[(size_t)sm[r] * D + cch];
            atomicAdd(&numb[(size_t)dm[r] * D + cch], ex * (vg + de[r]));
            atomicAdd(&denb[(size_t)dm[r] * D + cch], ex);
        }
    }
}

// ---------------- softmax state init / layer finalize ----------------
__global__ void init_softmax(float* __restrict__ mbuf, float* __restrict__ numb,
                             float* __restrict__ denb) {
    size_t i = (size_t)blockIdx.x * blockDim.x + threadIdx.x;
    mbuf[i] = -__builtin_inff();
    numb[i] = 0.0f;
    denb[i] = 0.0f;
}

__global__ void finalize_layer(float* __restrict__ numb, float* __restrict__ denb,
                               float* __restrict__ hout, float* __restrict__ mbuf) {
    size_t i = (size_t)blockIdx.x * blockDim.x + threadIdx.x;
    hout[i] = numb[i] / denb[i];           // every node has a self-loop -> den > 0
    mbuf[i] = -__builtin_inff();
    numb[i] = 0.0f;
    denb[i] = 0.0f;
}

// ---------------- global max pool + linear head ----------------
__global__ void pool_head(const float* __restrict__ h,
                          const float* __restrict__ outW, const float* __restrict__ outB,
                          float* __restrict__ out) {
    __shared__ float part[4][64];
    __shared__ float g[64];
    int b = blockIdx.x;                 // 32 graphs, contiguous 2048-node blocks
    int t = threadIdx.x;                // 256 threads
    int c = t & 63;
    int rg = t >> 6;                    // 0..3
    const int npg = N_NODES / NB;       // 2048
    float mx = -__builtin_inff();
    for (int i = rg; i < npg; i += 4)
        mx = fmaxf(mx, h[(size_t)(b * npg + i) * D + c]);
    part[rg][c] = mx;
    __syncthreads();
    if (t < 64)
        g[c] = fmaxf(fmaxf(part[0][c], part[1][c]), fmaxf(part[2][c], part[3][c]));
    __syncthreads();
    if (t < NC) {
        float acc = outB[t];
        for (int cc = 0; cc < D; ++cc) acc += g[cc] * outW[cc * NC + t];
        out[b * NC + t] = acc;
    }
}

// ---------------- launch ----------------
extern "C" void kernel_launch(void* const* d_in, const int* in_sizes, int n_in,
                              void* d_out, int out_size, void* d_ws, size_t ws_size,
                              hipStream_t stream) {
    const float* x       = (const float*)d_in[0];
    const float* pos     = (const float*)d_in[1];
    const float* W_lin   = (const float*)d_in[2];
    const float* W_src   = (const float*)d_in[3];
    const float* W_dst   = (const float*)d_in[4];
    const float* pos_W1  = (const float*)d_in[5];
    const float* pos_b1  = (const float*)d_in[6];
    const float* pos_W2  = (const float*)d_in[7];
    const float* pos_b2  = (const float*)d_in[8];
    const float* attn_W1 = (const float*)d_in[9];
    const float* attn_b1 = (const float*)d_in[10];
    const float* attn_W2 = (const float*)d_in[11];
    const float* attn_b2 = (const float*)d_in[12];
    const float* out_W   = (const float*)d_in[13];
    const float* out_b   = (const float*)d_in[14];
    const int*   eidx    = (const int*)d_in[15];
    const int*   srcI    = eidx;
    const int*   dstI    = eidx + N_EDGES;

    // workspace carve-out
    char* ws = (char*)d_ws;
    size_t off = 0;
    auto carve = [&](size_t bytes) -> void* {
        void* pt = ws + off;
        off = (off + bytes + 255) & ~(size_t)255;
        return pt;
    };
    const size_t WSZ = (size_t)NL * D * D;      // 16384 elems per weight family
    _Float16* Wl16  = (_Float16*)carve(WSZ * 2);
    _Float16* Ws16  = (_Float16*)carve(WSZ * 2);
    _Float16* Wd16  = (_Float16*)carve(WSZ * 2);
    _Float16* pW216 = (_Float16*)carve(WSZ * 2);
    _Float16* aW116 = (_Float16*)carve(WSZ * 2);
    _Float16* aW216 = (_Float16*)carve(WSZ * 2);
    const size_t NDB = (size_t)N_NODES * D * 4; // 16 MB node buffers
    float* vbuf   = (float*)carve(NDB);
    float* asbuf  = (float*)carve(NDB);
    float* adbuf  = (float*)carve(NDB);
    float* mbuf   = (float*)carve(NDB);
    float* numb   = (float*)carve(NDB);
    float* denb   = (float*)carve(NDB);
    float* hbuf0  = (float*)carve(NDB);
    float* hbuf1  = (float*)carve(NDB);
    const size_t EDB = (size_t)N_EDGES * D * 4; // 256 MB edge buffers
    float* deltaWs = (float*)carve(EDB);
    float* alphaWs = (float*)carve(EDB);
    (void)ws_size; (void)in_sizes; (void)n_in; (void)out_size;

    // convert + swizzle weights to f16 B-fragment layout
    {
        int n = (int)WSZ, blk = 256, grd = (n + blk - 1) / blk;
        swizzle_weights<<<grd, blk, 0, stream>>>(W_lin,   Wl16);
        swizzle_weights<<<grd, blk, 0, stream>>>(W_src,   Ws16);
        swizzle_weights<<<grd, blk, 0, stream>>>(W_dst,   Wd16);
        swizzle_weights<<<grd, blk, 0, stream>>>(pos_W2,  pW216);
        swizzle_weights<<<grd, blk, 0, stream>>>(attn_W1, aW116);
        swizzle_weights<<<grd, blk, 0, stream>>>(attn_W2, aW216);
    }
    // init softmax state
    init_softmax<<<(N_NODES * D) / 256, 256, 0, stream>>>(mbuf, numb, denb);

    const float* hin = x;
    float* hping[2] = { hbuf0, hbuf1 };
    for (int L = 0; L < NL; ++L) {
        float* hout = hping[L & 1];
        node_gemm<<<(N_NODES / 16) / 8, 256, 0, stream>>>(
            hin, Wl16 + (size_t)L * D * D, Ws16 + (size_t)L * D * D, Wd16 + (size_t)L * D * D,
            vbuf, asbuf, adbuf);
        edge_pass_b<<<(N_EDGES / 16) / 8, 256, 0, stream>>>(
            srcI, dstI, pos, asbuf, adbuf,
            pos_W1 + (size_t)L * 2 * D, pos_b1 + (size_t)L * D,
            pW216  + (size_t)L * D * D, pos_b2 + (size_t)L * D,
            aW116  + (size_t)L * D * D, attn_b1 + (size_t)L * D,
            aW216  + (size_t)L * D * D, attn_b2 + (size_t)L * D,
            deltaWs, alphaWs, mbuf);
        edge_pass_c<<<(N_EDGES / 16) / 8, 256, 0, stream>>>(
            srcI, dstI, vbuf, mbuf, deltaWs, alphaWs, numb, denb);
        finalize_layer<<<(N_NODES * D) / 256, 256, 0, stream>>>(numb, denb, hout, mbuf);
        hin = hout;
    }
    pool_head<<<NB, 256, 0, stream>>>(hin, out_W, out_b, (float*)d_out);
}
